// Net_LSTM_85435489452398
// MI455X (gfx1250) — matrix operand order repaired and checked by reading, not compiled, not run on hardware
//
#include <hip/hip_runtime.h>

// ---------------- problem constants ----------------
#define T_STEPS 100
#define BATCH   4096
#define CIN     14
#define HID     128
#define NCLS    7
#define GATES   512            // 4*HID
#define NTILES  32             // GATES/16
#define KT_H    4              // HID/32 (K-tiles of 32 for f16 WMMA)
#define ROWS    32             // batch rows per block
#define MT      2              // M-tiles (16 rows each) per block
#define THREADS 256            // 8 wave32 waves
#define NWAVES  8

typedef __attribute__((ext_vector_type(8)))  _Float16 v8h;
typedef __attribute__((ext_vector_type(16))) _Float16 v16h;
typedef __attribute__((ext_vector_type(8)))  float    v8f;

// global (address_space(1)) pointer types for weight-fragment loads
typedef const __attribute__((address_space(1))) v16h* gv16h_ptr;
typedef const __attribute__((address_space(1))) char* gchar_ptr;

__device__ __forceinline__ float sigm(float x)      { return 1.0f / (1.0f + __expf(-x)); }
__device__ __forceinline__ float tanh_fast(float x) { return 1.0f - 2.0f / (__expf(2.0f * x) + 1.0f); }

__device__ __forceinline__ v8f wmma_f16(v16h a, v16h b, v8f c) {
  // D = A(16x32 f16) * B(32x16 f16) + C(16x16 f32)
  return __builtin_amdgcn_wmma_f32_16x16x32_f16(false, a, false, b, (short)0, c, false, false);
}

// Build a 16x32 f16 A-fragment from a row-major LDS array.
// Documented 16-bit A layout: lane m = lane&15, half = lane>>4,
// element e -> k_local = ((e&8)<<1) | (half<<3) | (e&7)
// => two contiguous 8-half (16B) chunks at k = half*8 and k = 16 + half*8.
__device__ __forceinline__ v16h ldA(const _Float16* base, int stride, int rowBase,
                                    int kb, int m, int hw) {
  const _Float16* rp = base + (rowBase + m) * stride + kb + hw * 8;
  v8h lo = *(const v8h*)(rp);
  v8h hi = *(const v8h*)(rp + 16);
  return __builtin_shufflevector(lo, hi, 0,1,2,3,4,5,6,7,8,9,10,11,12,13,14,15);
}

// B-fragment load: clean uniform base pointer (SGPR pair, usable as saddr) +
// per-lane 32-bit voffset (made opaque per timestep by the caller's empty asm,
// which defeats LICM without dirtying the base) + compile-time immediate.
// Lowers to: global_load_b128 v, vOff, s[base] offset:IMM  (GVS form, LOADcnt
// only, no per-load address VALU -> no WMMA->VALU hazard NOPs).
__device__ __forceinline__ v16h ldB(const _Float16* base, unsigned int vOff,
                                    unsigned int imm) {
  gchar_ptr p = (gchar_ptr)base;
  return *(gv16h_ptr)(p + vOff + imm);
}

// ---------------- prepack: f32 weights -> f16 B-fragments ----------------
// B matrix (32x16, 16-bit) layout: same k order as A fragments,
// k_local = ((e&8)<<1) | (half<<3) | (e&7); N = lane&15.
// dst[(nt*ktiles + kt)*512 + lane*16 + e] = W[n][k], n = nt*16 + (lane&15),
// k = kt*32 + ((e&8)<<1) + (lane>>4)*8 + (e&7) (zero-padded past Kdim).
__global__ void prepack_bfrag(const float* __restrict__ W, _Float16* __restrict__ dst,
                              int Kdim, int ktiles) {
  int tid  = blockIdx.x * blockDim.x + threadIdx.x;
  int e    = tid & 15;
  int lane = (tid >> 4) & 31;
  int frag = tid >> 9;
  int kt   = frag % ktiles;
  int nt   = frag / ktiles;
  if (nt >= NTILES) return;
  int n = nt * 16 + (lane & 15);
  int k = kt * 32 + ((e & 8) << 1) + (lane >> 4) * 8 + (e & 7);
  float v = (k < Kdim) ? W[n * Kdim + k] : 0.0f;
  dst[(size_t)frag * 512 + lane * 16 + e] = (_Float16)v;
}

// ---------------- persistent 2-layer SLSTM kernel ----------------
__global__ __launch_bounds__(THREADS, 1)
void slstm_persistent(const float* __restrict__ x,
                      const _Float16* __restrict__ fWih1,
                      const _Float16* __restrict__ fWhh1,
                      const _Float16* __restrict__ fWih2,
                      const _Float16* __restrict__ fWhh2,
                      const float* __restrict__ bih1, const float* __restrict__ bhh1,
                      const float* __restrict__ bih2, const float* __restrict__ bhh2,
                      const float* __restrict__ thr1p, const float* __restrict__ thr2p,
                      const float* __restrict__ Wfc, const float* __restrict__ bfc,
                      float* __restrict__ out) {
  __shared__ __align__(16) _Float16 xpad[ROWS][32];      // x_t tile, K padded 14->32
  __shared__ __align__(16) _Float16 hMem1[ROWS][HID];    // layer1 membrane (recurrent h)
  __shared__ __align__(16) _Float16 hSpk1[ROWS][HID];    // layer1 spikes (layer2 input)
  __shared__ __align__(16) _Float16 hMem2[ROWS][HID];    // layer2 membrane
  __shared__ __align__(16) float    meanBuf[ROWS][HID];  // time-mean of mem2

  const int tid     = threadIdx.x;
  const int lane    = tid & 31;
  const int wv      = tid >> 5;     // wave id 0..7 -> hidden slice [16wv, 16wv+16)
  const int m       = lane & 15;
  const int hw      = lane >> 4;
  const int rowBase = blockIdx.x * ROWS;

  const float thr1 = thr1p[0];
  const float thr2 = thr2p[0];

  // 32-bit per-lane voffsets for the fragment arrays
  // byteOff = frag*1024 + lane*32 ; frag = (wv+8s)*ktiles + kt
  unsigned int vOff4 = (unsigned)(wv * KT_H) * 1024u + (unsigned)lane * 32u; // ktiles==4
  unsigned int vOff1 = (unsigned)wv * 1024u + (unsigned)lane * 32u;          // ktiles==1

  // zero-init recurrent LDS state
  for (int i = tid; i < ROWS * HID; i += THREADS) {
    (&hMem1[0][0])[i] = (_Float16)0.0f;
    (&hMem2[0][0])[i] = (_Float16)0.0f;
  }

  // per-lane gate biases (bih+bhh), constant over time
  float biasL1[4], biasL2[4];
#pragma unroll
  for (int s = 0; s < 4; ++s) {
    int n = 16 * (wv + 8 * s) + m;
    biasL1[s] = bih1[n] + bhh1[n];
    biasL2[s] = bih2[n] + bhh2[n];
  }

  // per-wave register state in C/D layout: [mt][vgpr r] -> row mt*16+r+8*hw, col 16*wv+m
  float syn1[MT][8], mem1[MT][8], syn2[MT][8], mem2[MT][8], m2sum[MT][8];
#pragma unroll
  for (int mt = 0; mt < MT; ++mt)
#pragma unroll
    for (int r = 0; r < 8; ++r) {
      syn1[mt][r] = 0.0f; mem1[mt][r] = 0.0f;
      syn2[mt][r] = 0.0f; mem2[mt][r] = 0.0f;
      m2sum[mt][r] = 0.0f;
    }

  // warm L2/L0 for layer-2 weight fragments (global_prefetch_b8)
  __builtin_prefetch(fWih2 + (size_t)(wv * KT_H) * 512 + (size_t)lane * 16, 0, 3);
  __builtin_prefetch(fWhh2 + (size_t)(wv * KT_H) * 512 + (size_t)lane * 16, 0, 3);

#pragma unroll 1
  for (int t = 0; t < T_STEPS; ++t) {
    // Make the per-lane voffsets opaque each timestep: the weight-fragment
    // loads (loop-invariant, L2-resident) then cannot be hoisted out of the
    // loop and spilled to scratch, while the bases stay clean uniform
    // pointers so the backend emits saddr-form global_load_b128.
    asm volatile("" : "+v"(vOff4), "+v"(vOff1));

    // ---- stage x_t tile into padded f16 LDS ----
    for (int i = tid; i < ROWS * 32; i += THREADS) {
      int r = i >> 5, c = i & 31;
      float v = (c < CIN) ? x[((size_t)t * BATCH + rowBase + r) * CIN + c] : 0.0f;
      xpad[r][c] = (_Float16)v;
    }
    __syncthreads();  // B0: xpad ready; prev-step hMem1/hMem2 writes visible

    // ---- layer 1 GEMM: gates = x@Wih1^T + mem1@Whh1^T + b ----
    v8f acc[MT][4];
#pragma unroll
    for (int mt = 0; mt < MT; ++mt)
#pragma unroll
      for (int s = 0; s < 4; ++s)
#pragma unroll
        for (int r = 0; r < 8; ++r) acc[mt][s][r] = biasL1[s];

    {
      v16h ax[MT];
#pragma unroll
      for (int mt = 0; mt < MT; ++mt) ax[mt] = ldA(&xpad[0][0], 32, mt * 16, 0, m, hw);
#pragma unroll
      for (int s = 0; s < 4; ++s) {
        // frag = wv + 8s  (ktiles==1) -> imm = s*8*1024
        v16h b = ldB(fWih1, vOff1, (unsigned)s * 8192u);
#pragma unroll
        for (int mt = 0; mt < MT; ++mt) acc[mt][s] = wmma_f16(ax[mt], b, acc[mt][s]);
      }
    }
#pragma unroll
    for (int kt = 0; kt < KT_H; ++kt) {
      v16h am[MT];
#pragma unroll
      for (int mt = 0; mt < MT; ++mt) am[mt] = ldA(&hMem1[0][0], HID, mt * 16, kt * 32, m, hw);
#pragma unroll
      for (int s = 0; s < 4; ++s) {
        // frag = (wv+8s)*4 + kt -> imm = s*32*1024 + kt*1024
        v16h b = ldB(fWhh1, vOff4, (unsigned)s * 32768u + (unsigned)kt * 1024u);
#pragma unroll
        for (int mt = 0; mt < MT; ++mt) acc[mt][s] = wmma_f16(am[mt], b, acc[mt][s]);
      }
    }
    __syncthreads();  // B1: everyone done reading hMem1/xpad

    // ---- layer 1 pointwise (i,f,g,o all wave-local) ----
#pragma unroll
    for (int mt = 0; mt < MT; ++mt) {
#pragma unroll
      for (int r = 0; r < 8; ++r) {
        float ig = sigm(acc[mt][0][r]);
        float fg = sigm(acc[mt][1][r]);
        float gg = tanh_fast(acc[mt][2][r]);
        float og = sigm(acc[mt][3][r]);
        float c2 = fg * syn1[mt][r] + ig * gg;
        float h  = og * tanh_fast(c2);
        float rst = (mem1[mt][r] - thr1 > 0.0f) ? thr1 : 0.0f;  // reset*thr
        float mn  = h - rst;
        float sp  = (mn - thr1 > 0.0f) ? 1.0f : 0.0f;
        syn1[mt][r] = c2;
        mem1[mt][r] = mn;
        int row = mt * 16 + r + hw * 8;
        int col = 16 * wv + m;
        hMem1[row][col] = (_Float16)mn;
        hSpk1[row][col] = (_Float16)sp;
      }
    }
    __syncthreads();  // B2: spk1/mem1 published

    // ---- layer 2 GEMM: gates = spk1@Wih2^T + mem2@Whh2^T + b ----
#pragma unroll
    for (int mt = 0; mt < MT; ++mt)
#pragma unroll
      for (int s = 0; s < 4; ++s)
#pragma unroll
        for (int r = 0; r < 8; ++r) acc[mt][s][r] = biasL2[s];

#pragma unroll
    for (int kt = 0; kt < KT_H; ++kt) {
      v16h as_[MT], am2[MT];
#pragma unroll
      for (int mt = 0; mt < MT; ++mt) {
        as_[mt] = ldA(&hSpk1[0][0], HID, mt * 16, kt * 32, m, hw);
        am2[mt] = ldA(&hMem2[0][0], HID, mt * 16, kt * 32, m, hw);
      }
#pragma unroll
      for (int s = 0; s < 4; ++s) {
        unsigned int imm = (unsigned)s * 32768u + (unsigned)kt * 1024u;
        v16h bi = ldB(fWih2, vOff4, imm);
#pragma unroll
        for (int mt = 0; mt < MT; ++mt) acc[mt][s] = wmma_f16(as_[mt], bi, acc[mt][s]);
        v16h bh = ldB(fWhh2, vOff4, imm);
#pragma unroll
        for (int mt = 0; mt < MT; ++mt) acc[mt][s] = wmma_f16(am2[mt], bh, acc[mt][s]);
      }
    }
    __syncthreads();  // B3: everyone done reading hSpk1/hMem2

    // ---- layer 2 pointwise ----
#pragma unroll
    for (int mt = 0; mt < MT; ++mt) {
#pragma unroll
      for (int r = 0; r < 8; ++r) {
        float ig = sigm(acc[mt][0][r]);
        float fg = sigm(acc[mt][1][r]);
        float gg = tanh_fast(acc[mt][2][r]);
        float og = sigm(acc[mt][3][r]);
        float c2 = fg * syn2[mt][r] + ig * gg;
        float h  = og * tanh_fast(c2);
        float rst = (mem2[mt][r] - thr2 > 0.0f) ? thr2 : 0.0f;
        float mn  = h - rst;
        syn2[mt][r] = c2;
        mem2[mt][r] = mn;
        m2sum[mt][r] += mn;
        int row = mt * 16 + r + hw * 8;
        int col = 16 * wv + m;
        hMem2[row][col] = (_Float16)mn;
      }
    }
    // next-iteration B0 orders hMem2 writes vs. reads
  }

  // ---- epilogue: mean over time, then FC to NCLS ----
#pragma unroll
  for (int mt = 0; mt < MT; ++mt)
#pragma unroll
    for (int r = 0; r < 8; ++r) {
      int row = mt * 16 + r + hw * 8;
      int col = 16 * wv + m;
      meanBuf[row][col] = m2sum[mt][r] * (1.0f / (float)T_STEPS);
    }
  __syncthreads();

  for (int i = tid; i < ROWS * NCLS; i += THREADS) {
    int r  = i / NCLS;
    int nc = i % NCLS;
    float s = bfc[nc];
#pragma unroll 4
    for (int j = 0; j < HID; ++j) s += meanBuf[r][j] * Wfc[nc * HID + j];
    out[(size_t)(rowBase + r) * NCLS + nc] = s;
  }
}

// ---------------- host-side launch ----------------
extern "C" void kernel_launch(void* const* d_in, const int* in_sizes, int n_in,
                              void* d_out, int out_size, void* d_ws, size_t ws_size,
                              hipStream_t stream) {
  (void)in_sizes; (void)n_in; (void)out_size; (void)ws_size;
  const float* x    = (const float*)d_in[0];
  const float* Wih1 = (const float*)d_in[1];
  const float* Whh1 = (const float*)d_in[2];
  const float* bih1 = (const float*)d_in[3];
  const float* bhh1 = (const float*)d_in[4];
  const float* thr1 = (const float*)d_in[5];
  const float* Wih2 = (const float*)d_in[6];
  const float* Whh2 = (const float*)d_in[7];
  const float* bih2 = (const float*)d_in[8];
  const float* bhh2 = (const float*)d_in[9];
  const float* thr2 = (const float*)d_in[10];
  const float* Wfc  = (const float*)d_in[11];
  const float* bfc  = (const float*)d_in[12];

  // workspace layout (f16 fragment arrays, 416 KB total)
  char* ws = (char*)d_ws;
  _Float16* fWhh1 = (_Float16*)(ws);                               // 32*4*512*2 = 128KB
  _Float16* fWih1 = (_Float16*)(ws + 131072);                      // 32*1*512*2 =  32KB
  _Float16* fWih2 = (_Float16*)(ws + 131072 + 32768);              // 128KB
  _Float16* fWhh2 = (_Float16*)(ws + 131072 + 32768 + 131072);     // 128KB

  prepack_bfrag<<<(NTILES * KT_H * 512) / 256, 256, 0, stream>>>(Whh1, fWhh1, HID, KT_H);
  prepack_bfrag<<<(NTILES * 1    * 512) / 256, 256, 0, stream>>>(Wih1, fWih1, CIN, 1);
  prepack_bfrag<<<(NTILES * KT_H * 512) / 256, 256, 0, stream>>>(Wih2, fWih2, HID, KT_H);
  prepack_bfrag<<<(NTILES * KT_H * 512) / 256, 256, 0, stream>>>(Whh2, fWhh2, HID, KT_H);

  slstm_persistent<<<BATCH / ROWS, THREADS, 0, stream>>>(
      x, fWih1, fWhh1, fWih2, fWhh2,
      bih1, bhh1, bih2, bhh2, thr1, thr2, Wfc, bfc, (float*)d_out);
}
